// SemanticExtractor_34119220199672
// MI455X (gfx1250) — matrix operand order repaired and checked by reading, not compiled, not run on hardware
//
#include <hip/hip_runtime.h>
#include <hip/hip_bf16.h>

// ---------------- problem constants ----------------
#define BB 16
#define MM 2048
#define DD 512
#define QQ 16
#define CHUNK 64
#define NCH (MM / CHUNK)   // 32

constexpr float INV_SQRT_D = 0.04419417382415922f;      // 1/sqrt(512)
constexpr float PE_C      = 9.210340371976184f / 512.f; // ln(10000)/D

typedef __attribute__((ext_vector_type(16))) _Float16 v16h;
typedef __attribute__((ext_vector_type(8)))  float    v8f;
typedef __attribute__((ext_vector_type(4)))  unsigned int u32x4;

union Frag { v16h v; u32x4 u[2]; };

// ---------------- kernel 0a: tiny projections ----------------
// Qk = query @ Wk  (f16), Kq = query @ Wk^T + bk, Vq = query @ Wv^T + bv
__global__ void __launch_bounds__(256) k_proj(
    const float* __restrict__ query, const float* __restrict__ Wk,
    const float* __restrict__ bk,    const float* __restrict__ Wv,
    const float* __restrict__ bv,
    _Float16* __restrict__ Qkh, float* __restrict__ Kq, float* __restrict__ Vq) {
  int t = blockIdx.x * 256 + threadIdx.x;       // 0..8191
  int q = t >> 9, d = t & 511;
  const float* qr = query + q * DD;
  float aQk = 0.f, aKq = bk[d], aVq = bv[d];
  for (int j = 0; j < DD; ++j) {
    float qv = qr[j];
    aQk += qv * Wk[j * DD + d];   // (query @ Wk)[q,d]
    aKq += qv * Wk[d * DD + j];   // (query @ Wk^T)[q,d]
    aVq += qv * Wv[d * DD + j];   // (query @ Wv^T)[q,d]
  }
  Qkh[t] = (_Float16)aQk;
  Kq[t]  = aKq;
  Vq[t]  = aVq;
}

// ---------------- kernel 0b: convert Wv to f16 ----------------
__global__ void __launch_bounds__(256) k_cvt(const float* __restrict__ Wv,
                                             _Float16* __restrict__ Wvh) {
  int t = blockIdx.x * 256 + threadIdx.x;
  if (t < DD * DD) Wvh[t] = (_Float16)Wv[t];
}

// ---------------- kernel 0c: s0[q] = query[q].Kq[q], qbk[q] = query[q].bk ----
__global__ void __launch_bounds__(256) k_s0(const float* __restrict__ query,
                                            const float* __restrict__ Kq,
                                            const float* __restrict__ bk,
                                            float* __restrict__ s0,
                                            float* __restrict__ qbk) {
  __shared__ float r1[16][16], r2[16][16];
  int t = threadIdx.x, q = t >> 4, seg = t & 15;
  float a = 0.f, b2 = 0.f;
  for (int d = seg * 32; d < seg * 32 + 32; ++d) {
    float qv = query[q * DD + d];
    a  += qv * Kq[q * DD + d];
    b2 += qv * bk[d];
  }
  r1[q][seg] = a; r2[q][seg] = b2;
  __syncthreads();
  if (t < 16) {
    float s = 0.f, sb = 0.f;
    for (int i = 0; i < 16; ++i) { s += r1[t][i]; sb += r2[t][i]; }
    s0[t] = s; qbk[t] = sb;
  }
}

// ---------------- kernel 1: main flash pass over patch ----------------
// grid (NCH, B), 256 threads (8 wave32)
__global__ void __launch_bounds__(256) k_main(
    const float* __restrict__ patch, const _Float16* __restrict__ Qkh,
    const float* __restrict__ qbk,
    float* __restrict__ Mpart, float* __restrict__ Lpart,
    _Float16* __restrict__ Ypart) {
  __shared__ __align__(16) _Float16 xrow[CHUNK][536];  // x+PE, m-major (A of GEMM1)
  __shared__ __align__(16) _Float16 xT[DD][72];        // transposed (B of GEMM2)
  __shared__ float spb[2][16][68];                     // raw scores, per K-half
  __shared__ __align__(16) _Float16 pbh[16][72];       // exp(scores - max) (A of GEMM2)
  __shared__ float red[16][16];
  __shared__ float cmax[16];

  const int c = blockIdx.x, b = blockIdx.y;
  const int tid  = threadIdx.x;
  const int wave = tid >> 5, lane = tid & 31, r = lane & 15, hh = lane >> 4;
  const float* xs = patch + ((size_t)b * MM + (size_t)c * CHUNK) * DD;
  const int posbase = c * CHUNK;

  // ---- stage x = patch + sinusoidal PE into LDS (f16), both layouts ----
  // Each thread owns one float4 column (d4 loop-invariant -> PE freqs hoisted)
  {
    const int d4 = (tid & 127) << 2;
    const int mstart = tid >> 7;                 // 0 or 1
    const float f0 = __expf(-(float)d4 * PE_C);
    const float f1 = __expf(-(float)(d4 + 2) * PE_C);
    for (int m = mstart; m < CHUNK; m += 2) {
      float4 v = *(const float4*)(xs + (size_t)m * DD + d4);
      float pos = (float)(posbase + m);
      float sn0, cs0, sn1, cs1;
      __sincosf(pos * f0, &sn0, &cs0);
      __sincosf(pos * f1, &sn1, &cs1);
      _Float16 h0 = (_Float16)(v.x + sn0), h1 = (_Float16)(v.y + cs0);
      _Float16 h2 = (_Float16)(v.z + sn1), h3 = (_Float16)(v.w + cs1);
      _Float16* xr = &xrow[m][d4];
      xr[0] = h0; xr[1] = h1; xr[2] = h2; xr[3] = h3;
      xT[d4 + 0][m] = h0; xT[d4 + 1][m] = h1;
      xT[d4 + 2][m] = h2; xT[d4 + 3][m] = h3;
    }
  }
  __syncthreads();

  // ---- phase A: sp[m,q] = x @ Qk^T via WMMA; 4 m-tiles x 2 K-halves = 8 waves
  {
    const int mt = wave & 3, kb = (wave >> 2) << 8;   // K base 0 or 256
    const int m0 = mt << 4;
    v8f acc = {0,0,0,0,0,0,0,0};
    const _Float16* ar = &xrow[m0 + r][0];
    const _Float16* br = Qkh + (size_t)r * DD;
    for (int kc = kb; kc < kb + 256; kc += 32) {
      Frag a, bf2;
      a.u[0]   = *(const u32x4*)(ar + kc + hh * 8);
      a.u[1]   = *(const u32x4*)(ar + kc + 16 + hh * 8);
      bf2.u[0] = *(const u32x4*)(br + kc + hh * 8);
      bf2.u[1] = *(const u32x4*)(br + kc + 16 + hh * 8);
      acc = __builtin_amdgcn_wmma_f32_16x16x32_f16(
          false, a.v, false, bf2.v, (short)0, acc, false, false);
    }
    float* dst = &spb[wave >> 2][r][0];   // N = q = r ; rows M = m
#pragma unroll
    for (int i = 0; i < 8; ++i) dst[m0 + (hh << 3) + i] = acc[i];
  }
  __syncthreads();

  // ---- phase B: chunk-local online softmax partials ----
  {
    const int q = tid & 15, row = tid >> 4;
    const float qb = qbk[q];
    float vals[4]; float mloc = -3.0e38f;
#pragma unroll
    for (int j = 0; j < 4; ++j) {
      int mm2 = row + (j << 4);
      float s = (spb[0][q][mm2] + spb[1][q][mm2] + qb) * INV_SQRT_D;
      vals[j] = s; mloc = fmaxf(mloc, s);
    }
    red[row][q] = mloc;
    __syncthreads();
    if (tid < 16) {
      float g = red[0][tid];
      for (int i = 1; i < 16; ++i) g = fmaxf(g, red[i][tid]);
      cmax[tid] = g;
    }
    __syncthreads();
    float g = cmax[q], lsum = 0.f;
#pragma unroll
    for (int j = 0; j < 4; ++j) {
      int mm2 = row + (j << 4);
      float e = __expf(vals[j] - g);
      lsum += e;
      pbh[q][mm2] = (_Float16)e;
    }
    red[row][q] = lsum;
    __syncthreads();
    if (tid < 16) {
      float L = 0.f;
      for (int i = 0; i < 16; ++i) L += red[i][tid];
      int o = (b * NCH + c) * 16 + tid;
      Mpart[o] = cmax[tid];
      Lpart[o] = L;
    }
  }
  __syncthreads();

  // ---- phase C: y_part[q,d] = p @ x via WMMA (K = m = 64), stored f16 ----
  {
    const _Float16* ap = &pbh[r][0];     // A rows = q
    for (int j = 0; j < 4; ++j) {
      int dt = wave + (j << 3);
      v8f acc = {0,0,0,0,0,0,0,0};
      const _Float16* bp = &xT[dt * 16 + r][0];   // B cols = d
#pragma unroll
      for (int mc = 0; mc < CHUNK; mc += 32) {
        Frag a, bb;
        a.u[0]  = *(const u32x4*)(ap + mc + hh * 8);
        a.u[1]  = *(const u32x4*)(ap + mc + 16 + hh * 8);
        bb.u[0] = *(const u32x4*)(bp + mc + hh * 8);
        bb.u[1] = *(const u32x4*)(bp + mc + 16 + hh * 8);
        acc = __builtin_amdgcn_wmma_f32_16x16x32_f16(
            false, a.v, false, bb.v, (short)0, acc, false, false);
      }
      _Float16* yp = Ypart + (size_t)((b * NCH + c) * 16) * DD + dt * 16 + r;
#pragma unroll
      for (int i = 0; i < 8; ++i) yp[(size_t)(i + (hh << 3)) * DD] = (_Float16)acc[i];
    }
  }
}

// ---------------- kernel 2: combine chunk partials -> normalized y (f16), w0
__global__ void __launch_bounds__(256) k_combine(
    const float* __restrict__ s0, const float* __restrict__ Mpart,
    const float* __restrict__ Lpart, const _Float16* __restrict__ Ypart,
    _Float16* __restrict__ Yh, float* __restrict__ W0) {
  __shared__ float scc[NCH][16];
  __shared__ float Linv[16];
  int b = blockIdx.x, t = threadIdx.x;
  if (t < 16) {
    float s0s = s0[t] * INV_SQRT_D;
    float g = s0s;
    for (int c2 = 0; c2 < NCH; ++c2) g = fmaxf(g, Mpart[(b * NCH + c2) * 16 + t]);
    float e0 = __expf(s0s - g), L = e0;
    for (int c2 = 0; c2 < NCH; ++c2) {
      float sc = __expf(Mpart[(b * NCH + c2) * 16 + t] - g);
      scc[c2][t] = sc;
      L += Lpart[(b * NCH + c2) * 16 + t] * sc;
    }
    Linv[t] = 1.0f / L;
    W0[b * 16 + t] = e0 / L;
  }
  __syncthreads();
  for (int idx = t; idx < 16 * DD; idx += 256) {
    int q = idx >> 9, d = idx & 511;
    float acc = 0.f;
    for (int c2 = 0; c2 < NCH; ++c2)
      acc += scc[c2][q] * (float)Ypart[((size_t)(b * NCH + c2) * 16 + q) * DD + d];
    Yh[(size_t)(b * 16 + q) * DD + d] = (_Float16)(acc * Linv[q]);
  }
}

// ---------------- kernel 3: patch_token = w0*Vq + y @ Wv^T + (1-w0)*bv (WMMA)
__global__ void __launch_bounds__(256) k_pt(
    const _Float16* __restrict__ Yh, const _Float16* __restrict__ Wvh,
    const float* __restrict__ Vq, const float* __restrict__ bv,
    const float* __restrict__ W0, float* __restrict__ PT) {
  int b = blockIdx.x, tid = threadIdx.x;
  int wave = tid >> 5, lane = tid & 31, r = lane & 15, hh = lane >> 4;
  const _Float16* ar = Yh + (size_t)(b * 16 + r) * DD;
  for (int j = 0; j < 4; ++j) {
    int dt = wave + (j << 3);
    int d  = dt * 16 + r;
    const _Float16* br = Wvh + (size_t)d * DD;   // B[k][n=d] = Wv[d,k] -> contiguous
    v8f acc = {0,0,0,0,0,0,0,0};
    for (int kc = 0; kc < DD; kc += 32) {
      Frag a, bb;
      a.u[0]  = *(const u32x4*)(ar + kc + hh * 8);
      a.u[1]  = *(const u32x4*)(ar + kc + 16 + hh * 8);
      bb.u[0] = *(const u32x4*)(br + kc + hh * 8);
      bb.u[1] = *(const u32x4*)(br + kc + 16 + hh * 8);
      acc = __builtin_amdgcn_wmma_f32_16x16x32_f16(
          false, a.v, false, bb.v, (short)0, acc, false, false);
    }
    float bvd = bv[d];
#pragma unroll
    for (int i = 0; i < 8; ++i) {
      int q = i + (hh << 3);
      float w0v = W0[b * 16 + q];
      PT[(size_t)(b * 16 + q) * DD + d] = acc[i] + w0v * Vq[q * DD + d] + (1.0f - w0v) * bvd;
    }
  }
}

// ---------------- kernel 4: fuse softmax over Q and final reduction -------
__global__ void __launch_bounds__(256) k_fuse(
    const float* __restrict__ PT, const float* __restrict__ Wf,
    const float* __restrict__ bf, float* __restrict__ out) {
  __shared__ float red2[16][16];
  __shared__ float ftv[16];
  __shared__ float fw[16];
  int b = blockIdx.x, t = threadIdx.x;
  int q = t >> 4, seg = t & 15;
  float p2 = 0.f;
  for (int d = seg * 32; d < seg * 32 + 32; ++d)
    p2 += PT[(size_t)(b * 16 + q) * DD + d] * Wf[d];
  red2[q][seg] = p2;
  __syncthreads();
  if (t < 16) {
    float s = bf[0];
    for (int i = 0; i < 16; ++i) s += red2[t][i];
    ftv[t] = s;
  }
  __syncthreads();
  if (t < 16) {
    float mx = ftv[0];
    for (int i = 1; i < 16; ++i) mx = fmaxf(mx, ftv[i]);
    float sm = 0.f;
    for (int i = 0; i < 16; ++i) sm += __expf(ftv[i] - mx);
    fw[t] = __expf(ftv[t] - mx) / sm;
  }
  __syncthreads();
  for (int d = t; d < DD; d += 256) {
    float acc = 0.f;
    for (int qq = 0; qq < 16; ++qq)
      acc += PT[(size_t)(b * 16 + qq) * DD + d] * fw[qq];
    out[b * DD + d] = acc;
  }
}

// ---------------- host launcher ----------------
extern "C" void kernel_launch(void* const* d_in, const int* in_sizes, int n_in,
                              void* d_out, int out_size, void* d_ws, size_t ws_size,
                              hipStream_t stream) {
  const float* patch = (const float*)d_in[0];
  const float* query = (const float*)d_in[1];
  const float* Wk    = (const float*)d_in[2];
  const float* bk    = (const float*)d_in[3];
  const float* Wv    = (const float*)d_in[4];
  const float* bv    = (const float*)d_in[5];
  const float* Wf    = (const float*)d_in[6];
  const float* bf    = (const float*)d_in[7];
  float* out = (float*)d_out;

  char* w = (char*)d_ws;
  size_t o = 0;
  auto take = [&](size_t bytes) { char* p = w + o; o = (o + bytes + 255) & ~(size_t)255; return p; };
  _Float16* Qkh  = (_Float16*)take(QQ * DD * 2);
  float*    Kq   = (float*)   take(QQ * DD * 4);
  float*    Vq   = (float*)   take(QQ * DD * 4);
  _Float16* Wvh  = (_Float16*)take(DD * DD * 2);
  float*    s0   = (float*)   take(QQ * 4);
  float*    qbk  = (float*)   take(QQ * 4);
  float*    Mp   = (float*)   take((size_t)BB * NCH * 16 * 4);
  float*    Lp   = (float*)   take((size_t)BB * NCH * 16 * 4);
  _Float16* Yp   = (_Float16*)take((size_t)BB * NCH * 16 * DD * 2);
  _Float16* Yh   = (_Float16*)take((size_t)BB * 16 * DD * 2);
  float*    W0   = (float*)   take((size_t)BB * 16 * 4);
  float*    PT   = (float*)   take((size_t)BB * 16 * DD * 4);
  (void)ws_size; (void)in_sizes; (void)n_in; (void)out_size;

  k_proj<<<32, 256, 0, stream>>>(query, Wk, bk, Wv, bv, Qkh, Kq, Vq);
  k_cvt<<<(DD * DD + 255) / 256, 256, 0, stream>>>(Wv, Wvh);
  k_s0<<<1, 256, 0, stream>>>(query, Kq, bk, s0, qbk);
  k_main<<<dim3(NCH, BB), 256, 0, stream>>>(patch, Qkh, qbk, Mp, Lp, Yp);
  k_combine<<<BB, 256, 0, stream>>>(s0, Mp, Lp, Yp, Yh, W0);
  k_pt<<<BB, 256, 0, stream>>>(Yh, Wvh, Vq, bv, W0, PT);
  k_fuse<<<BB, 256, 0, stream>>>(PT, Wf, bf, out);
}
